// GraphAttentionLayer_5617817223458
// MI455X (gfx1250) — matrix operand order repaired and checked by reading, not compiled, not run on hardware
//
#include <hip/hip_runtime.h>

typedef __bf16 bf16_t;
typedef __attribute__((ext_vector_type(16))) __bf16 v16bf;
typedef __attribute__((ext_vector_type(8)))  __bf16 v8bf;
typedef __attribute__((ext_vector_type(8)))  float   v8f;

#define NB    4
#define NN    4096
#define FIN   256
#define DOUT  128
#define SCALE 0.08838834764831845f   /* 1/sqrt(128) */
#define NEGBIG (-9.0e15f)

#if defined(__has_builtin)
# if __has_builtin(__builtin_amdgcn_global_load_async_to_lds_b128)
#  define HAVE_ASYNC 1
# endif
#endif
#ifndef HAVE_ASYNC
# define HAVE_ASYNC 0
#endif

// The async builtin wants pointers to GCC-style int4 vectors in AS1/AS3
// (from the round-2 diagnostic: "__vector_size__(4*sizeof(int)) int __device__ *").
typedef int v4i_vs __attribute__((vector_size(16)));
typedef __attribute__((address_space(1))) v4i_vs as1_v4i;
typedef __attribute__((address_space(3))) v4i_vs as3_v4i;

static __device__ __forceinline__ v16bf cat8(v8bf lo, v8bf hi) {
  v16bf r;
#pragma unroll
  for (int i = 0; i < 8; ++i) { r[i] = lo[i]; r[i + 8] = hi[i]; }
  return r;
}
static __device__ __forceinline__ v16bf ld_pair(const bf16_t* p0, const bf16_t* p1) {
  return cat8(*(const v8bf*)p0, *(const v8bf*)p1);
}
// reductions across the 16 lanes of each wave half (xor masks 1..8 stay in-half)
static __device__ __forceinline__ float half_max(float v) {
#pragma unroll
  for (int off = 8; off; off >>= 1) v = fmaxf(v, __shfl_xor(v, off, 32));
  return v;
}
static __device__ __forceinline__ float half_sum(float v) {
#pragma unroll
  for (int off = 8; off; off >>= 1) v += __shfl_xor(v, off, 32);
  return v;
}

// 16-byte global -> LDS copy: async engine if available, sync fallback.
static __device__ __forceinline__ void cp16B(bf16_t* l, const bf16_t* g) {
#if HAVE_ASYNC
  __builtin_amdgcn_global_load_async_to_lds_b128((as1_v4i*)g, (as3_v4i*)l, 0, 0);
#else
  *(v8bf*)l = *(const v8bf*)g;
#endif
}
template <int N>
static __device__ __forceinline__ void wait_async() {
#if HAVE_ASYNC
# if __has_builtin(__builtin_amdgcn_s_wait_asynccnt)
  __builtin_amdgcn_s_wait_asynccnt(N);
# else
  if (N == 4) asm volatile("s_wait_asynccnt 0x4" ::: "memory");
  else        asm volatile("s_wait_asynccnt 0x0" ::: "memory");
# endif
#endif
}

// ---------- Kernel 1: f32 -> bf16 conversions ----------
__global__ void __launch_bounds__(256) gat_cvt(
    const float* __restrict__ h,  const float* __restrict__ wq,
    const float* __restrict__ wk, const float* __restrict__ wv,
    bf16_t* __restrict__ hb, bf16_t* __restrict__ wqb,
    bf16_t* __restrict__ wkb, bf16_t* __restrict__ wvb) {
  int i = blockIdx.x * blockDim.x + threadIdx.x;
  if (i < NB * NN * FIN) hb[i] = (bf16_t)h[i];
  if (i < DOUT * FIN) {
    wqb[i] = (bf16_t)wq[i];
    wkb[i] = (bf16_t)wk[i];
    wvb[i] = (bf16_t)wv[i];
  }
}

// ---------- Kernel 2: QKV projection via WMMA ----------
// One wave computes a 16x16 tile of (h @ W^T + bias). V is stored transposed.
__global__ void __launch_bounds__(256) gat_proj(
    const bf16_t* __restrict__ hb,
    const bf16_t* __restrict__ wqb, const bf16_t* __restrict__ wkb,
    const bf16_t* __restrict__ wvb,
    const float* __restrict__ bq, const float* __restrict__ bk,
    const float* __restrict__ bv,
    bf16_t* __restrict__ qo, bf16_t* __restrict__ ko, bf16_t* __restrict__ vto) {
  const int lane = threadIdx.x & 31;
  const int wave = threadIdx.x >> 5;
  int u = blockIdx.x * 8 + wave;          // 24576 wave units
  const int j  = u & 7;  u >>= 3;         // output d-chunk (0..7)
  const int t  = u % 3;                   // 0=Q 1=K 2=V
  const int rb = u / 3;                   // row block (0..1023)
  const int g  = lane >> 4;
  const int ln = lane & 15;

  const bf16_t* w   = (t == 0) ? wqb : (t == 1) ? wkb : wvb;
  const float*  bia = (t == 0) ? bq  : (t == 1) ? bk  : bv;
  const bf16_t* arow = hb + (size_t)(rb * 16 + ln) * FIN;  // A: row = lane M
  const bf16_t* brow = w  + (size_t)(j * 16 + ln) * FIN;   // B: col = lane N

  v8f acc = {};
#pragma unroll
  for (int c = 0; c < 8; ++c) {           // K = 256 in 8 steps of 32
    v16bf a = ld_pair(arow + 32 * c + 16 * g, arow + 32 * c + 16 * g + 8);
    v16bf b = ld_pair(brow + 32 * c + 16 * g, brow + 32 * c + 16 * g + 8);
    acc = __builtin_amdgcn_wmma_f32_16x16x32_bf16(false, a, false, b,
                                                  (short)0, acc, false, false);
  }
  const float bval = bia[j * 16 + ln];
  if (t < 2) {
    bf16_t* outp = (t == 0) ? qo : ko;
#pragma unroll
    for (int r = 0; r < 8; ++r)
      outp[(size_t)(rb * 16 + r + 8 * g) * DOUT + j * 16 + ln] =
          (bf16_t)(acc[r] + bval);
  } else {
#pragma unroll
    for (int r = 0; r < 8; ++r) {
      int grow = rb * 16 + r + 8 * g;
      int bi = grow >> 12, nn = grow & (NN - 1);
      vto[((size_t)bi * DOUT + j * 16 + ln) * NN + nn] = (bf16_t)(acc[r] + bval);
    }
  }
}

// ---------- Kernel 3: masked attention, online softmax, WMMA ----------
// Block = 8 waves = 8 query blocks of the same batch; all share the K/V tile
// stream, staged once per block into double-buffered LDS via async copies.
__global__ void __launch_bounds__(256) gat_attn(
    const bf16_t* __restrict__ qb, const bf16_t* __restrict__ kb,
    const bf16_t* __restrict__ vt, const int* __restrict__ adj,
    float* __restrict__ out) {
  __shared__ __align__(16) bf16_t ksh[2][32][136];   // padded pitch: 272B
  __shared__ __align__(16) bf16_t vsh[2][128][40];   // padded pitch: 80B
  __shared__ __align__(16) bf16_t pbuf[8][16][40];   // per-wave P staging

  const int tid  = threadIdx.x;
  const int lane = tid & 31;
  const int wave = tid >> 5;
  const int gq   = blockIdx.x * 8 + wave;  // 0..1023 query blocks
  const int bi   = gq >> 8;
  const int qloc = (gq & 255) << 4;
  const int g    = lane >> 4;
  const int ln   = lane & 15;

  const bf16_t* kbase_p = kb + (size_t)bi * NN * DOUT;
  const bf16_t* vbase_p = vt + (size_t)bi * DOUT * NN;
  const int*    abase_p = adj + ((size_t)bi << 24);

  // staging slices: K tile 32x128 (8KB), V tile 128x32 (8KB), 256 threads
  const int kr = tid >> 3, kc = (tid & 7) << 4;  // K: row, 16-elem column chunk
  const int vd = tid >> 1, vc = (tid & 1) << 4;  // V: row, 16-elem column chunk
  auto stage_in = [&](int itn) {
    const int st = itn & 1;
    const bf16_t* kg = kbase_p + ((size_t)itn << 5) * DOUT;  // contiguous rows
    const bf16_t* vg = vbase_p + (itn << 5);
    cp16B(&ksh[st][kr][kc],     kg + (size_t)kr * DOUT + kc);
    cp16B(&ksh[st][kr][kc + 8], kg + (size_t)kr * DOUT + kc + 8);
    cp16B(&vsh[st][vd][vc],     vg + (size_t)vd * NN + vc);
    cp16B(&vsh[st][vd][vc + 8], vg + (size_t)vd * NN + vc + 8);
  };

  // preload Q as four 16x32 A-tiles (d = 128)
  const bf16_t* qrow = qb + (size_t)(bi * NN + qloc + ln) * DOUT;
  v16bf qa[4];
#pragma unroll
  for (int c = 0; c < 4; ++c)
    qa[c] = ld_pair(qrow + 32 * c + 16 * g, qrow + 32 * c + 16 * g + 8);

  v8f o[8] = {};
  float rmax[8], rsum[8];
#pragma unroll
  for (int r = 0; r < 8; ++r) { rmax[r] = -3.0e38f; rsum[r] = 0.f; }

  stage_in(0);
  for (int it = 0; it < 128; ++it) {
    const int st = it & 1;
    const int k0 = it << 5;
    if (it < 127) { stage_in(it + 1); wait_async<4>(); }  // in-order: stage it done
    else          { wait_async<0>(); }
    __syncthreads();

    // S = Q @ K^T for 32 keys (two 16x16 tiles, K-dim 128 in 4 chunks)
    v8f s0 = {}, s1 = {};
#pragma unroll
    for (int c = 0; c < 4; ++c) {
      v16bf b0 = ld_pair(&ksh[st][ln][32 * c + 16 * g],
                         &ksh[st][ln][32 * c + 16 * g + 8]);
      v16bf b1 = ld_pair(&ksh[st][16 + ln][32 * c + 16 * g],
                         &ksh[st][16 + ln][32 * c + 16 * g + 8]);
      s0 = __builtin_amdgcn_wmma_f32_16x16x32_bf16(false, qa[c], false, b0,
                                                   (short)0, s0, false, false);
      s1 = __builtin_amdgcn_wmma_f32_16x16x32_bf16(false, qa[c], false, b1,
                                                   (short)0, s1, false, false);
    }
    // mask + online softmax update (row = r + 8g, col = ln)
    float f[8];
#pragma unroll
    for (int r = 0; r < 8; ++r) {
      const int* arow = abase_p + (size_t)(qloc + r + 8 * g) * NN + k0 + ln;
      int a0 = arow[0];
      int a1 = arow[16];
      if (it < 127) __builtin_prefetch(arow + 32, 0, 1);
      float v0 = (a0 > 0) ? s0[r] * SCALE : NEGBIG;
      float v1 = (a1 > 0) ? s1[r] * SCALE : NEGBIG;
      float m  = half_max(fmaxf(v0, v1));
      float nm = fmaxf(rmax[r], m);
      f[r] = __expf(rmax[r] - nm);
      rmax[r] = nm;
      float e0 = __expf(v0 - nm);
      float e1 = __expf(v1 - nm);
      rsum[r] = rsum[r] * f[r] + half_sum(e0 + e1);
      pbuf[wave][r + 8 * g][ln]      = (bf16_t)e0;
      pbuf[wave][r + 8 * g][16 + ln] = (bf16_t)e1;
    }
    // rescale accumulator
#pragma unroll
    for (int dc = 0; dc < 8; ++dc)
#pragma unroll
      for (int r = 0; r < 8; ++r) o[dc][r] *= f[r];
    // make P visible before re-reading it in A-layout (LDS in-order per wave)
    asm volatile("s_wait_dscnt 0" ::: "memory");
    v16bf pa = ld_pair(&pbuf[wave][ln][16 * g], &pbuf[wave][ln][16 * g + 8]);
    // O += P @ V  (B tiles from LDS-staged transposed V)
#pragma unroll
    for (int dc = 0; dc < 8; ++dc) {
      v16bf bv = ld_pair(&vsh[st][dc * 16 + ln][16 * g],
                         &vsh[st][dc * 16 + ln][16 * g + 8]);
      o[dc] = __builtin_amdgcn_wmma_f32_16x16x32_bf16(false, pa, false, bv,
                                                      (short)0, o[dc], false, false);
    }
    __syncthreads();  // protect buffers before next stage overwrites
  }
  // epilogue: divide by softmax denominators, write f32 output
#pragma unroll
  for (int r = 0; r < 8; ++r) rsum[r] = 1.0f / rsum[r];
#pragma unroll
  for (int dc = 0; dc < 8; ++dc)
#pragma unroll
    for (int r = 0; r < 8; ++r)
      out[(size_t)(bi * NN + qloc + r + 8 * g) * DOUT + dc * 16 + ln] =
          o[dc][r] * rsum[r];
}

extern "C" void kernel_launch(void* const* d_in, const int* in_sizes, int n_in,
                              void* d_out, int out_size, void* d_ws, size_t ws_size,
                              hipStream_t stream) {
  const float* h   = (const float*)d_in[0];
  const int*   adj = (const int*)d_in[1];
  const float* Wq  = (const float*)d_in[2];
  const float* bq  = (const float*)d_in[3];
  const float* Wk  = (const float*)d_in[4];
  const float* bk  = (const float*)d_in[5];
  const float* Wv  = (const float*)d_in[6];
  const float* bv  = (const float*)d_in[7];
  float* out = (float*)d_out;

  // workspace carve (bf16 elements, all 16B aligned): ~21 MB total
  bf16_t* hb  = (bf16_t*)d_ws;                 // 16384 x 256
  bf16_t* wqb = hb  + (size_t)NB * NN * FIN;   // 128 x 256
  bf16_t* wkb = wqb + (size_t)DOUT * FIN;
  bf16_t* wvb = wkb + (size_t)DOUT * FIN;
  bf16_t* qo  = wvb + (size_t)DOUT * FIN;      // 16384 x 128
  bf16_t* ko  = qo  + (size_t)NB * NN * DOUT;
  bf16_t* vto = ko  + (size_t)NB * NN * DOUT;  // [B][128][4096]

  gat_cvt<<<(NB * NN * FIN + 255) / 256, 256, 0, stream>>>(
      h, Wq, Wk, Wv, hb, wqb, wkb, wvb);
  gat_proj<<<3072, 256, 0, stream>>>(
      hb, wqb, wkb, wvb, bq, bk, bv, qo, ko, vto);
  gat_attn<<<128, 256, 0, stream>>>(qo, ko, vto, adj, out);
}
